// CellFiltering_32031866093751
// MI455X (gfx1250) — matrix operand order repaired and checked by reading, not compiled, not run on hardware
//
#include <hip/hip_runtime.h>

#define DIM_L 256
#define N_CTX 1024
#define NB_ROWS 65536
#define N_REC 8
#define BATCH_B 8192

typedef __attribute__((ext_vector_type(16))) _Float16 v16h;
typedef __attribute__((ext_vector_type(8)))  _Float16 v8h;
typedef __attribute__((ext_vector_type(8)))  float    v8f;

// ---------------------------------------------------------------------------
// Fragment loaders. f32-source versions convert in-register (fallback path);
// f16-source versions are pure loads (preferred path, pre-converted operands).
// ---------------------------------------------------------------------------
__device__ __forceinline__ v16h pack16(float4 f0, float4 f1, float4 f2, float4 f3) {
    v16h a;
    a[0]  = (_Float16)f0.x; a[1]  = (_Float16)f0.y; a[2]  = (_Float16)f0.z; a[3]  = (_Float16)f0.w;
    a[4]  = (_Float16)f1.x; a[5]  = (_Float16)f1.y; a[6]  = (_Float16)f1.z; a[7]  = (_Float16)f1.w;
    a[8]  = (_Float16)f2.x; a[9]  = (_Float16)f2.y; a[10] = (_Float16)f2.z; a[11] = (_Float16)f2.w;
    a[12] = (_Float16)f3.x; a[13] = (_Float16)f3.y; a[14] = (_Float16)f3.z; a[15] = (_Float16)f3.w;
    return a;
}

// A fragment (16x32 f16): lane holds M=lane&15, K = koff+0..15
__device__ __forceinline__ v16h load_a_frag(const float* rowp, int koff) {
    return pack16(*(const float4*)(rowp + koff),
                  *(const float4*)(rowp + koff + 4),
                  *(const float4*)(rowp + koff + 8),
                  *(const float4*)(rowp + koff + 12));
}
__device__ __forceinline__ v16h load_a_frag(const _Float16* rowp, int koff) {
    v8h lo = *(const v8h*)(rowp + koff);
    v8h hi = *(const v8h*)(rowp + koff + 8);
    return __builtin_shufflevector(lo, hi, 0,1,2,3,4,5,6,7,8,9,10,11,12,13,14,15);
}

// B fragment (32x16 f16): lane holds column N=lane&15, 16 contiguous K values
__device__ __forceinline__ v16h load_b_frag(const float* p) {
    return pack16(*(const float4*)(p),     *(const float4*)(p + 4),
                  *(const float4*)(p + 8), *(const float4*)(p + 12));
}
__device__ __forceinline__ v16h load_b_frag(const _Float16* p) {
    v8h lo = *(const v8h*)(p);
    v8h hi = *(const v8h*)(p + 8);
    return __builtin_shufflevector(lo, hi, 0,1,2,3,4,5,6,7,8,9,10,11,12,13,14,15);
}

// ---------------------------------------------------------------------------
// f32 -> f16 bulk converts, 8 elements per thread. Split variant emits
// hi = rnd16(v) and lo = rnd16(v - hi) for near-f32 WMMA GEMMs.
// ---------------------------------------------------------------------------
__global__ void __launch_bounds__(256) cvt_f16_kernel(const float* __restrict__ src,
                                                      _Float16* __restrict__ dst, int n8) {
    int i = blockIdx.x * 256 + threadIdx.x;
    if (i >= n8) return;
    const float4* s = (const float4*)src + (size_t)i * 2;
    float4 a = s[0], b = s[1];
    v8h o;
    o[0] = (_Float16)a.x; o[1] = (_Float16)a.y; o[2] = (_Float16)a.z; o[3] = (_Float16)a.w;
    o[4] = (_Float16)b.x; o[5] = (_Float16)b.y; o[6] = (_Float16)b.z; o[7] = (_Float16)b.w;
    *((v8h*)dst + i) = o;
}

__global__ void __launch_bounds__(256) cvt_split_kernel(const float* __restrict__ src,
                                                        _Float16* __restrict__ dhi,
                                                        _Float16* __restrict__ dlo, int n8) {
    int i = blockIdx.x * 256 + threadIdx.x;
    if (i >= n8) return;
    const float4* s = (const float4*)src + (size_t)i * 2;
    float4 a = s[0], b = s[1];
    float v[8] = {a.x, a.y, a.z, a.w, b.x, b.y, b.z, b.w};
    v8h h, l;
#pragma unroll
    for (int j = 0; j < 8; ++j) {
        _Float16 hh = (_Float16)v[j];
        h[j] = hh;
        l[j] = (_Float16)(v[j] - (float)hh);
    }
    ((v8h*)dhi)[i] = h;
    ((v8h*)dlo)[i] = l;
}

// ---------------------------------------------------------------------------
// Kernel 0: inverse context norms. 1 wave per context row.
// ---------------------------------------------------------------------------
__global__ void __launch_bounds__(32) norms_kernel(const float* __restrict__ ctx,
                                                   float* __restrict__ inv_cn) {
    int c = blockIdx.x;
    int lane = threadIdx.x;
    const float* p = ctx + (size_t)c * DIM_L + lane * 8;
    float4 a = *(const float4*)p;
    float4 b = *(const float4*)(p + 4);
    float s = a.x*a.x + a.y*a.y + a.z*a.z + a.w*a.w
            + b.x*b.x + b.y*b.y + b.z*b.z + b.w*b.w;
#pragma unroll
    for (int m = 1; m < 32; m <<= 1) s += __shfl_xor(s, m, 32);
    if (lane == 0) inv_cn[c] = rsqrtf(fmaxf(s, 1e-24f));
}

// ---------------------------------------------------------------------------
// Kernel 1: per-row argmax over 1024 contexts of dot(x_i, ctx_c)/||ctx_c||.
// One wave per 32-row (2x16) tile: two A tiles register-resident, streamed B
// double-buffered so the loadcnt wait overlaps the previous chunk's WMMAs.
// Strict '>' in the tile loop = first-max-wins (cb ascends); tie-break index
// only in the final cross-lane reduction.
// ---------------------------------------------------------------------------
template <typename T>
__global__ void __launch_bounds__(128) argmax_kernel(const T* __restrict__ x,
                                                     const T* __restrict__ ctx,
                                                     const float* __restrict__ inv_cn,
                                                     int* __restrict__ argm) {
    int lane = threadIdx.x & 31;
    int wave = threadIdx.x >> 5;
    int R    = (blockIdx.x * 4 + wave) * 32;

    int M    = lane & 15;
    int half = lane >> 4;
    int akoff = half * 16;
    int bkoff = half * 16;
    int N    = M;

    const T* arow0 = x + (size_t)(R + M) * DIM_L;
    const T* arow1 = x + (size_t)(R + 16 + M) * DIM_L;
    v16h A0[8], A1[8];
#pragma unroll
    for (int kc = 0; kc < 8; ++kc) {
        A0[kc] = load_a_frag(arow0 + kc * 32, akoff);
        A1[kc] = load_a_frag(arow1 + kc * 32, akoff);
    }

    float bv0[8], bv1[8]; int bi0[8], bi1[8];
#pragma unroll
    for (int j = 0; j < 8; ++j) { bv0[j] = -3.4e38f; bi0[j] = 0; bv1[j] = -3.4e38f; bi1[j] = 0; }

    for (int cb = 0; cb < N_CTX; cb += 16) {
        const T* brow = ctx + (size_t)(cb + N) * DIM_L + bkoff;
        v8f acc0 = {0.f,0.f,0.f,0.f,0.f,0.f,0.f,0.f};
        v8f acc1 = {0.f,0.f,0.f,0.f,0.f,0.f,0.f,0.f};
        v16h B = load_b_frag(brow);
#pragma unroll
        for (int kc = 0; kc < 8; ++kc) {
            v16h Bn = B;
            if (kc < 7) Bn = load_b_frag(brow + (kc + 1) * 32);
            acc0 = __builtin_amdgcn_wmma_f32_16x16x32_f16(false, A0[kc], false, B,
                                                          (short)0, acc0, false, false);
            acc1 = __builtin_amdgcn_wmma_f32_16x16x32_f16(false, A1[kc], false, B,
                                                          (short)0, acc1, false, false);
            B = Bn;
        }
        float ic = inv_cn[cb + N];
        int idx = cb + N;
#pragma unroll
        for (int j = 0; j < 8; ++j) {
            float v0 = acc0[j] * ic;
            if (v0 > bv0[j]) { bv0[j] = v0; bi0[j] = idx; }
            float v1 = acc1[j] * ic;
            if (v1 > bv1[j]) { bv1[j] = v1; bi1[j] = idx; }
        }
    }

#pragma unroll
    for (int j = 0; j < 8; ++j) {
#pragma unroll
        for (int m = 1; m < 16; m <<= 1) {
            float ov = __shfl_xor(bv0[j], m, 32);
            int   oi = __shfl_xor(bi0[j], m, 32);
            if (ov > bv0[j] || (ov == bv0[j] && oi < bi0[j])) { bv0[j] = ov; bi0[j] = oi; }
            ov = __shfl_xor(bv1[j], m, 32);
            oi = __shfl_xor(bi1[j], m, 32);
            if (ov > bv1[j] || (ov == bv1[j] && oi < bi1[j])) { bv1[j] = ov; bi1[j] = oi; }
        }
        if (N == 0) {
            argm[R + half * 8 + j]      = bi0[j];
            argm[R + 16 + half * 8 + j] = bi1[j];
        }
    }
}

// ---------------------------------------------------------------------------
// Kernel 2: sequential EMA chain per context id (exact scan semantics).
// One wave per context; 256-d buffer register-resident. Scan batched 4x.
// ---------------------------------------------------------------------------
__global__ void __launch_bounds__(32) chain_kernel(const float* __restrict__ x,
                                                   const float* __restrict__ ctx,
                                                   const float* __restrict__ ctx_mod,
                                                   const int* __restrict__ argm,
                                                   float* __restrict__ act) {
    int c = blockIdx.x;
    int lane = threadIdx.x;

    float buf[8];
    {
        const float* cp = ctx + (size_t)c * DIM_L + lane * 8;
        float4 c0 = *(const float4*)cp;
        float4 c1 = *(const float4*)(cp + 4);
        buf[0] = c0.x; buf[1] = c0.y; buf[2] = c0.z; buf[3] = c0.w;
        buf[4] = c1.x; buf[5] = c1.y; buf[6] = c1.z; buf[7] = c1.w;
    }

    float cm[4][8];
#pragma unroll
    for (int s = 0; s < 4; ++s) {
        const float* mp = ctx_mod + (size_t)s * DIM_L + lane * 8;
        float4 m0 = *(const float4*)mp;
        float4 m1 = *(const float4*)(mp + 4);
        cm[s][0] = m0.x; cm[s][1] = m0.y; cm[s][2] = m0.z; cm[s][3] = m0.w;
        cm[s][4] = m1.x; cm[s][5] = m1.y; cm[s][6] = m1.z; cm[s][7] = m1.w;
    }

    const float invN = 1.0f / 50000.0f;

    for (int ib = 0; ib < NB_ROWS; ib += 128) {
        int id0 = argm[ib + lane];
        int id1 = argm[ib + 32 + lane];
        int id2 = argm[ib + 64 + lane];
        int id3 = argm[ib + 96 + lane];
        unsigned masks[4];
        masks[0] = __builtin_amdgcn_ballot_w32(id0 == c);
        masks[1] = __builtin_amdgcn_ballot_w32(id1 == c);
        masks[2] = __builtin_amdgcn_ballot_w32(id2 == c);
        masks[3] = __builtin_amdgcn_ballot_w32(id3 == c);
#pragma unroll
        for (int q = 0; q < 4; ++q) {
            unsigned mask = masks[q];
            int base = ib + q * 32;
            while (mask) {
                int b = __ffs(mask) - 1;
                mask &= mask - 1;
                int i = base + b;

                float p0 = 0.f, p1 = 0.f, p2 = 0.f, p3 = 0.f;
#pragma unroll
                for (int j = 0; j < 8; ++j) {
                    float bj = buf[j];
                    p0 += cm[0][j] * bj; p1 += cm[1][j] * bj;
                    p2 += cm[2][j] * bj; p3 += cm[3][j] * bj;
                }
#pragma unroll
                for (int m = 1; m < 32; m <<= 1) {
                    p0 += __shfl_xor(p0, m, 32);
                    p1 += __shfl_xor(p1, m, 32);
                    p2 += __shfl_xor(p2, m, 32);
                    p3 += __shfl_xor(p3, m, 32);
                }
                float mx = fmaxf(fmaxf(p0, p1), fmaxf(p2, p3));
                float a  = 1.0f / (1.0f + expf(-mx));
                if (lane == 0) act[i] = a;

                const float* xr = x + (size_t)i * DIM_L + lane * 8;
                float4 x0 = *(const float4*)xr;
                float4 x1 = *(const float4*)(xr + 4);
                buf[0] += (x0.x - buf[0]) * invN;
                buf[1] += (x0.y - buf[1]) * invN;
                buf[2] += (x0.z - buf[2]) * invN;
                buf[3] += (x0.w - buf[3]) * invN;
                buf[4] += (x1.x - buf[4]) * invN;
                buf[5] += (x1.y - buf[5]) * invN;
                buf[6] += (x1.z - buf[6]) * invN;
                buf[7] += (x1.w - buf[7]) * invN;
            }
        }
    }
}

// ---------------------------------------------------------------------------
// Kernel 3a (preferred): split-f16 receptor GEMM. x = xhi + xlo, W = whi + wlo;
// acc = xhi*whi + xhi*wlo + xlo*whi (f32 accumulate) ~= f32 GEMM to ~2^-22.
// One wave per (16-batch x 16-col) tile; W hi/lo register-resident (128 VGPRs);
// streamed A double-buffered.
// ---------------------------------------------------------------------------
__global__ void __launch_bounds__(128) receptor_split_kernel(const _Float16* __restrict__ xhi,
                                                             const _Float16* __restrict__ xlo,
                                                             const _Float16* __restrict__ whi,
                                                             const _Float16* __restrict__ wlo,
                                                             const float* __restrict__ bias,
                                                             const float* __restrict__ act,
                                                             float* __restrict__ out) {
    int lane = threadIdx.x & 31;
    int wave = threadIdx.x >> 5;
    int lbase = (blockIdx.y * 4 + wave) * 16;
    int bbase = blockIdx.x * 16;

    int M    = lane & 15;
    int half = lane >> 4;
    int akoff = half * 16;
    int N    = M;

    v16h Bh[8], Bl[8];
    const _Float16* wrh = whi + (size_t)(lbase + N) * DIM_L + half * 16;
    const _Float16* wrl = wlo + (size_t)(lbase + N) * DIM_L + half * 16;
#pragma unroll
    for (int kc = 0; kc < 8; ++kc) {
        Bh[kc] = load_b_frag(wrh + kc * 32);
        Bl[kc] = load_b_frag(wrl + kc * 32);
    }

    float bl = bias[lbase + N];

    v8f oacc = {0.f,0.f,0.f,0.f,0.f,0.f,0.f,0.f};
    for (int n = 0; n < N_REC; ++n) {
        const _Float16* ah = xhi + ((size_t)n * BATCH_B + bbase + M) * DIM_L;
        const _Float16* al = xlo + ((size_t)n * BATCH_B + bbase + M) * DIM_L;
        v8f acc = {0.f,0.f,0.f,0.f,0.f,0.f,0.f,0.f};
        v16h Ah = load_a_frag(ah, akoff);
        v16h Al = load_a_frag(al, akoff);
#pragma unroll
        for (int kc = 0; kc < 8; ++kc) {
            v16h AhN = Ah, AlN = Al;
            if (kc < 7) {
                AhN = load_a_frag(ah + (kc + 1) * 32, akoff);
                AlN = load_a_frag(al + (kc + 1) * 32, akoff);
            }
            acc = __builtin_amdgcn_wmma_f32_16x16x32_f16(false, Ah, false, Bh[kc],
                                                         (short)0, acc, false, false);
            acc = __builtin_amdgcn_wmma_f32_16x16x32_f16(false, Ah, false, Bl[kc],
                                                         (short)0, acc, false, false);
            acc = __builtin_amdgcn_wmma_f32_16x16x32_f16(false, Al, false, Bh[kc],
                                                         (short)0, acc, false, false);
            Ah = AhN; Al = AlN;
        }
#pragma unroll
        for (int j = 0; j < 8; ++j) {
            int bm = bbase + half * 8 + j;
            float a = act[n * BATCH_B + bm];
            float v = acc[j] + bl;
            oacc[j] += 0.5f * v * (1.0f + erff(v * 0.70710678118654752f)) * a;
        }
    }
#pragma unroll
    for (int j = 0; j < 8; ++j) {
        int bm = bbase + half * 8 + j;
        out[(size_t)bm * DIM_L + lbase + N] = oacc[j] * 0.125f;
    }
}

// ---------------------------------------------------------------------------
// Kernel 3b (fallback tiers): plain GEMM, two W tiles register-resident.
// ---------------------------------------------------------------------------
template <typename T>
__global__ void __launch_bounds__(128) receptor_kernel(const T* __restrict__ x,
                                                       const T* __restrict__ W,
                                                       const float* __restrict__ bias,
                                                       const float* __restrict__ act,
                                                       float* __restrict__ out) {
    int lane = threadIdx.x & 31;
    int wave = threadIdx.x >> 5;
    int lbase = (blockIdx.y * 4 + wave) * 32;
    int bbase = blockIdx.x * 16;

    int M    = lane & 15;
    int half = lane >> 4;
    int akoff = half * 16;
    int N    = M;

    v16h B0[8], B1[8];
    const T* wrow0 = W + (size_t)(lbase + N) * DIM_L + half * 16;
    const T* wrow1 = W + (size_t)(lbase + 16 + N) * DIM_L + half * 16;
#pragma unroll
    for (int kc = 0; kc < 8; ++kc) {
        B0[kc] = load_b_frag(wrow0 + kc * 32);
        B1[kc] = load_b_frag(wrow1 + kc * 32);
    }

    float bl0 = bias[lbase + N];
    float bl1 = bias[lbase + 16 + N];

    v8f o0 = {0.f,0.f,0.f,0.f,0.f,0.f,0.f,0.f};
    v8f o1 = {0.f,0.f,0.f,0.f,0.f,0.f,0.f,0.f};
    for (int n = 0; n < N_REC; ++n) {
        const T* arow = x + ((size_t)n * BATCH_B + bbase + M) * DIM_L;
        v8f acc0 = {0.f,0.f,0.f,0.f,0.f,0.f,0.f,0.f};
        v8f acc1 = {0.f,0.f,0.f,0.f,0.f,0.f,0.f,0.f};
        v16h A = load_a_frag(arow, akoff);
#pragma unroll
        for (int kc = 0; kc < 8; ++kc) {
            v16h An = A;
            if (kc < 7) An = load_a_frag(arow + (kc + 1) * 32, akoff);
            acc0 = __builtin_amdgcn_wmma_f32_16x16x32_f16(false, A, false, B0[kc],
                                                          (short)0, acc0, false, false);
            acc1 = __builtin_amdgcn_wmma_f32_16x16x32_f16(false, A, false, B1[kc],
                                                          (short)0, acc1, false, false);
            A = An;
        }
#pragma unroll
        for (int j = 0; j < 8; ++j) {
            int bm = bbase + half * 8 + j;
            float a = act[n * BATCH_B + bm];
            float v0 = acc0[j] + bl0;
            o0[j] += 0.5f * v0 * (1.0f + erff(v0 * 0.70710678118654752f)) * a;
            float v1 = acc1[j] + bl1;
            o1[j] += 0.5f * v1 * (1.0f + erff(v1 * 0.70710678118654752f)) * a;
        }
    }
#pragma unroll
    for (int j = 0; j < 8; ++j) {
        int bm = bbase + half * 8 + j;
        out[(size_t)bm * DIM_L + lbase + N]      = o0[j] * 0.125f;
        out[(size_t)bm * DIM_L + lbase + 16 + N] = o1[j] * 0.125f;
    }
}

// ---------------------------------------------------------------------------
extern "C" void kernel_launch(void* const* d_in, const int* in_sizes, int n_in,
                              void* d_out, int out_size, void* d_ws, size_t ws_size,
                              hipStream_t stream) {
    const float* x       = (const float*)d_in[0];   // (8, 8192, 256)
    const float* ctx     = (const float*)d_in[1];   // (1024, 256)
    const float* ctx_mod = (const float*)d_in[2];   // (4, 256)
    const float* W       = (const float*)d_in[3];   // (256, 256)
    const float* bias    = (const float*)d_in[4];   // (256,)
    float* out = (float*)d_out;                      // (8192, 256)

    char* ws = (char*)d_ws;
    const size_t sz_x16 = (size_t)NB_ROWS * DIM_L * 2;   // 32 MB
    const size_t sz_c16 = (size_t)N_CTX * DIM_L * 2;     // 512 KB
    const size_t sz_w16 = (size_t)DIM_L * DIM_L * 2;     // 128 KB

    const size_t off_argm = 0;                              // 65536 i32
    const size_t off_act  = off_argm + (size_t)NB_ROWS * 4; // 65536 f32
    const size_t off_inv  = off_act + (size_t)NB_ROWS * 4;  // 1024 f32
    const size_t off_base = off_inv + 4096;

    // Tier 1 (split): xhi, xlo, c16, whi, wlo
    const size_t off_xhi = off_base;
    const size_t off_xlo = off_xhi + sz_x16;
    const size_t off_c16s = off_xlo + sz_x16;
    const size_t off_whi = off_c16s + sz_c16;
    const size_t off_wlo = off_whi + sz_w16;
    const size_t need_split = off_wlo + sz_w16;
    // Tier 2 (plain f16): x16, c16, w16
    const size_t off_x16p = off_base;
    const size_t off_c16p = off_x16p + sz_x16;
    const size_t off_w16p = off_c16p + sz_c16;
    const size_t need_f16 = off_w16p + sz_w16;

    int*   argm   = (int*)(ws + off_argm);
    float* actb   = (float*)(ws + off_act);
    float* inv_cn = (float*)(ws + off_inv);

    const int n8_x = NB_ROWS * DIM_L / 8;
    const int n8_c = N_CTX * DIM_L / 8;
    const int n8_w = DIM_L * DIM_L / 8;

    norms_kernel<<<N_CTX, 32, 0, stream>>>(ctx, inv_cn);

    if (ws_size >= need_split) {
        _Float16* xhi = (_Float16*)(ws + off_xhi);
        _Float16* xlo = (_Float16*)(ws + off_xlo);
        _Float16* c16 = (_Float16*)(ws + off_c16s);
        _Float16* whi = (_Float16*)(ws + off_whi);
        _Float16* wlo = (_Float16*)(ws + off_wlo);
        cvt_split_kernel<<<(n8_x + 255) / 256, 256, 0, stream>>>(x, xhi, xlo, n8_x);
        cvt_f16_kernel<<<(n8_c + 255) / 256, 256, 0, stream>>>(ctx, c16, n8_c);
        cvt_split_kernel<<<(n8_w + 255) / 256, 256, 0, stream>>>(W, whi, wlo, n8_w);

        argmax_kernel<_Float16><<<NB_ROWS / 32 / 4, 128, 0, stream>>>(xhi, c16, inv_cn, argm);
        chain_kernel<<<N_CTX, 32, 0, stream>>>(x, ctx, ctx_mod, argm, actb);
        receptor_split_kernel<<<dim3(BATCH_B / 16, 4), 128, 0, stream>>>(xhi, xlo, whi, wlo,
                                                                         bias, actb, out);
    } else if (ws_size >= need_f16) {
        _Float16* x16 = (_Float16*)(ws + off_x16p);
        _Float16* c16 = (_Float16*)(ws + off_c16p);
        _Float16* w16 = (_Float16*)(ws + off_w16p);
        cvt_f16_kernel<<<(n8_x + 255) / 256, 256, 0, stream>>>(x, x16, n8_x);
        cvt_f16_kernel<<<(n8_c + 255) / 256, 256, 0, stream>>>(ctx, c16, n8_c);
        cvt_f16_kernel<<<(n8_w + 255) / 256, 256, 0, stream>>>(W, w16, n8_w);

        argmax_kernel<_Float16><<<NB_ROWS / 32 / 4, 128, 0, stream>>>(x16, c16, inv_cn, argm);
        chain_kernel<<<N_CTX, 32, 0, stream>>>(x, ctx, ctx_mod, argm, actb);
        receptor_kernel<_Float16><<<dim3(BATCH_B / 16, 2), 128, 0, stream>>>(x16, w16, bias, actb, out);
    } else {
        argmax_kernel<float><<<NB_ROWS / 32 / 4, 128, 0, stream>>>(x, ctx, inv_cn, argm);
        chain_kernel<<<N_CTX, 32, 0, stream>>>(x, ctx, ctx_mod, argm, actb);
        receptor_kernel<float><<<dim3(BATCH_B / 16, 2), 128, 0, stream>>>(x, W, bias, actb, out);
    }
}
